// NormAttention_24197845745831
// MI455X (gfx1250) — matrix operand order repaired and checked by reading, not compiled, not run on hardware
//
#include <hip/hip_runtime.h>
#include <hip/hip_bf16.h>

// MI455X (gfx1250, wave32) attention block.
// Pipeline: cvt/transpose -> WMMA GEMM(qkv,f16 out) -> norm+rope -> flash attn (WMMA) -> WMMA GEMM(proj,f32 out)
// GEMM stages tiles with GLOBAL_LOAD_ASYNC_TO_LDS_B128 (ASYNCcnt) and computes 32x32 per wave (4 WMMAs/k-step).

#define DIM     1024
#define NHEADS  16
#define HDIM    64
#define SEQ     1024
#define BATCH   8
#define ROWS    (BATCH * SEQ)   // 8192
#define QK_SCALE 0.125f         // 1/sqrt(64)

typedef _Float16 v2h  __attribute__((ext_vector_type(2)));
typedef _Float16 v8h  __attribute__((ext_vector_type(8)));
typedef _Float16 v16h __attribute__((ext_vector_type(16)));
typedef float    v8f  __attribute__((ext_vector_type(8)));

// A-fragment (16x32 f16): lane holds k chunks [p..p+7] and [p+16..p+23]
static __device__ __forceinline__ v16h ld_a16(const _Float16* p) {
    v8h lo = *(const v8h*)p;
    v8h hi = *(const v8h*)(p + 16);
    return __builtin_shufflevector(lo, hi, 0,1,2,3,4,5,6,7,8,9,10,11,12,13,14,15);
}
// B-fragment (32x16 f16): lane holds k chunks [p..p+7] and [p+16..p+23]
static __device__ __forceinline__ v16h ld_b16(const _Float16* p) {
    v8h lo = *(const v8h*)p;
    v8h hi = *(const v8h*)(p + 16);
    return __builtin_shufflevector(lo, hi, 0,1,2,3,4,5,6,7,8,9,10,11,12,13,14,15);
}
static __device__ __forceinline__ v8f wmma_f16(v16h a, v16h b, v8f c) {
    return __builtin_amdgcn_wmma_f32_16x16x32_f16(false, a, false, b, (short)0, c, false, false);
}
static __device__ __forceinline__ float wave_sum32(float x) {
    x += __shfl_xor(x, 1, 32);
    x += __shfl_xor(x, 2, 32);
    x += __shfl_xor(x, 4, 32);
    x += __shfl_xor(x, 8, 32);
    x += __shfl_xor(x, 16, 32);
    return x;
}

// LDS byte offset of a shared-memory pointer (AS(3) pointers are 32-bit LDS offsets)
static __device__ __forceinline__ unsigned lds_off(const void* p) {
    return (unsigned)(unsigned long long)(__attribute__((address_space(3))) const void*)p;
}
// Async DMA: 16 bytes global -> LDS per lane (tracked by ASYNCcnt)
static __device__ __forceinline__ void async_copy_b128(unsigned lds, const void* gptr) {
    asm volatile("global_load_async_to_lds_b128 %0, %1, off"
                 :: "v"(lds), "v"((unsigned long long)gptr)
                 : "memory");
}
static __device__ __forceinline__ void wait_async0() {
    asm volatile("s_wait_asynccnt 0" ::: "memory");
}

// ---------------- conversion kernels ----------------
__global__ __launch_bounds__(256) void na_cvt_f16(const float* __restrict__ s,
                                                  _Float16* __restrict__ d, int n) {
    for (int i = blockIdx.x * blockDim.x + threadIdx.x; i < n; i += gridDim.x * blockDim.x)
        d[i] = (_Float16)s[i];
}

// src [K,N] f32 -> dst [N,K] f16
__global__ __launch_bounds__(256) void na_transpose_f16(const float* __restrict__ src,
                                                        _Float16* __restrict__ dst,
                                                        int K, int N) {
    int idx = blockIdx.x * blockDim.x + threadIdx.x;
    int k = idx / N, n = idx - k * N;
    dst[(size_t)n * K + k] = (_Float16)src[idx];
}

// ---------------- WMMA GEMM: C[M,N] = A[M,K] @ Bt[N,K]^T + bias ----------------
// block tile 128x64, 8 waves (4 along M x 2 along N), 32x32 per wave, k-step 32
__global__ __launch_bounds__(256) void na_gemm_wmma(
    const _Float16* __restrict__ A,    // [M,K] f16
    const _Float16* __restrict__ Bt,   // [N,K] f16 (pre-transposed weights)
    const float*    __restrict__ bias, // [N]
    _Float16* __restrict__ Cf16,       // [M,N] f16 out (or null)
    float*    __restrict__ Cf32,       // [M,N] f32 out (or null)
    int M, int N, int K)
{
    __shared__ __align__(16) _Float16 As[128][40];
    __shared__ __align__(16) _Float16 Bs[64][40];

    const int n0 = blockIdx.x * 64;
    const int m0 = blockIdx.y * 128;
    const int tid  = threadIdx.x;
    const int wave = tid >> 5, lane = tid & 31;
    const int lm = lane & 15, lh = lane >> 4;
    const int wm = wave >> 1, wn = wave & 1;

    // stagers: A tile 128x32 halves (2 x b128 per thread), B tile 64x32 halves (1 x b128)
    const int arow = tid >> 1,  acol = (tid & 1) * 16;   // 0..127, {0,16}
    const int brow = tid >> 2,  bcol = (tid & 3) * 8;    // 0..63,  {0,8,16,24}
    const unsigned lAs0 = lds_off(&As[arow][acol]);
    const unsigned lAs1 = lds_off(&As[arow][acol + 8]);
    const unsigned lBs  = lds_off(&Bs[brow][bcol]);
    const _Float16* gA = &A [(size_t)(m0 + arow) * K + acol];
    const _Float16* gB = &Bt[(size_t)(n0 + brow) * K + bcol];

    v8f acc00 = {}, acc01 = {}, acc10 = {}, acc11 = {};

    for (int k0 = 0; k0 < K; k0 += 32) {
        async_copy_b128(lAs0, gA + k0);
        async_copy_b128(lAs1, gA + k0 + 8);
        async_copy_b128(lBs,  gB + k0);
        if (k0 + 32 < K) {
            __builtin_prefetch(gA + k0 + 32, 0, 1);
            __builtin_prefetch(gB + k0 + 32, 0, 1);
        }
        wait_async0();
        __syncthreads();

        v16h a0 = ld_a16(&As[wm * 32 +      lm][lh * 8]);
        v16h a1 = ld_a16(&As[wm * 32 + 16 + lm][lh * 8]);
        v16h b0 = ld_b16(&Bs[wn * 32 +      lm][lh * 8]);
        v16h b1 = ld_b16(&Bs[wn * 32 + 16 + lm][lh * 8]);
        acc00 = wmma_f16(a0, b0, acc00);
        acc01 = wmma_f16(a0, b1, acc01);
        acc10 = wmma_f16(a1, b0, acc10);
        acc11 = wmma_f16(a1, b1, acc11);
        __syncthreads();
    }

    const int c0 = n0 + wn * 32 + lm;
    const int c1 = c0 + 16;
    const float bv0 = bias[c0], bv1 = bias[c1];
    const int rb0 = m0 + wm * 32 + lh * 8;
    const int rb1 = rb0 + 16;
    if (Cf16) {
        #pragma unroll
        for (int r = 0; r < 8; ++r) {
            Cf16[(size_t)(rb0 + r) * N + c0] = (_Float16)(acc00[r] + bv0);
            Cf16[(size_t)(rb0 + r) * N + c1] = (_Float16)(acc01[r] + bv1);
            Cf16[(size_t)(rb1 + r) * N + c0] = (_Float16)(acc10[r] + bv0);
            Cf16[(size_t)(rb1 + r) * N + c1] = (_Float16)(acc11[r] + bv1);
        }
    } else {
        #pragma unroll
        for (int r = 0; r < 8; ++r) {
            Cf32[(size_t)(rb0 + r) * N + c0] = acc00[r] + bv0;
            Cf32[(size_t)(rb0 + r) * N + c1] = acc01[r] + bv1;
            Cf32[(size_t)(rb1 + r) * N + c0] = acc10[r] + bv0;
            Cf32[(size_t)(rb1 + r) * N + c1] = acc11[r] + bv1;
        }
    }
}

// ---------------- LayerNorm + RoPE (wave per (b,h,s) row of 64) ----------------
__global__ __launch_bounds__(256) void na_norm_rope(
    const _Float16* __restrict__ qkv,  // [ROWS, 3*DIM] f16
    const float* __restrict__ qn_w, const float* __restrict__ qn_b,
    const float* __restrict__ kn_w, const float* __restrict__ kn_b,
    const float* __restrict__ cosT, const float* __restrict__ sinT, // [SEQ, HDIM]
    _Float16* __restrict__ Qd,   // [B,H,S,D] (pre-scaled by 1/sqrt(D))
    _Float16* __restrict__ Kd,   // [B,H,S,D]
    _Float16* __restrict__ Vt)   // [B,H,D,S]
{
    const int w    = (blockIdx.x * blockDim.x + threadIdx.x) >> 5;  // 0..B*H*S-1
    const int lane = threadIdx.x & 31;
    const int b = w >> 14;              // H*S = 16384
    const int h = (w >> 10) & (NHEADS - 1);
    const int s = w & (SEQ - 1);
    const int d0 = lane * 2;

    const size_t rowoff = (size_t)(b * SEQ + s) * (3 * DIM) + h * HDIM + d0;
    v2h qv = *(const v2h*)&qkv[rowoff];
    v2h kv = *(const v2h*)&qkv[rowoff + DIM];
    v2h vv = *(const v2h*)&qkv[rowoff + 2 * DIM];

    float q0 = (float)qv[0], q1 = (float)qv[1];
    float k0 = (float)kv[0], k1 = (float)kv[1];

    const float inv64 = 1.0f / 64.0f;
    float qmu  = wave_sum32(q0 + q1) * inv64;
    float qsq  = wave_sum32(q0 * q0 + q1 * q1) * inv64;
    float qrs  = rsqrtf(qsq - qmu * qmu + 1e-6f);
    float kmu  = wave_sum32(k0 + k1) * inv64;
    float ksq  = wave_sum32(k0 * k0 + k1 * k1) * inv64;
    float krs  = rsqrtf(ksq - kmu * kmu + 1e-6f);

    float qw0 = qn_w[d0], qw1 = qn_w[d0 + 1], qb0 = qn_b[d0], qb1 = qn_b[d0 + 1];
    float kw0 = kn_w[d0], kw1 = kn_w[d0 + 1], kb0 = kn_b[d0], kb1 = kn_b[d0 + 1];
    float qn0 = (q0 - qmu) * qrs * qw0 + qb0;
    float qn1 = (q1 - qmu) * qrs * qw1 + qb1;
    float kn0 = (k0 - kmu) * krs * kw0 + kb0;
    float kn1 = (k1 - kmu) * krs * kw1 + kb1;

    float c0 = cosT[s * HDIM + d0], c1 = cosT[s * HDIM + d0 + 1];
    float s0 = sinT[s * HDIM + d0], s1 = sinT[s * HDIM + d0 + 1];
    // rotate_half: pair (a,b) -> (-b, a)
    float qr0 = (qn0 * c0 - qn1 * s0) * QK_SCALE;
    float qr1 = (qn1 * c1 + qn0 * s1) * QK_SCALE;
    float kr0 = kn0 * c0 - kn1 * s0;
    float kr1 = kn1 * c1 + kn0 * s1;

    const size_t qoff = ((size_t)((b * NHEADS + h) * SEQ + s)) * HDIM + d0;
    v2h qo; qo[0] = (_Float16)qr0; qo[1] = (_Float16)qr1;
    v2h ko; ko[0] = (_Float16)kr0; ko[1] = (_Float16)kr1;
    *(v2h*)&Qd[qoff] = qo;
    *(v2h*)&Kd[qoff] = ko;

    const size_t voff = ((size_t)((b * NHEADS + h) * HDIM + d0)) * SEQ + s;
    Vt[voff]       = vv[0];
    Vt[voff + SEQ] = vv[1];
}

// ---------------- flash attention (wave owns 16 queries; 32-key tiles) ----------------
__global__ __launch_bounds__(256) void na_attention(
    const _Float16* __restrict__ Q,   // [B*H, S, D] (pre-scaled)
    const _Float16* __restrict__ K,   // [B*H, S, D]
    const _Float16* __restrict__ Vt,  // [B*H, D, S]
    _Float16* __restrict__ Out)       // [ROWS, DIM] f16 (feeds proj GEMM)
{
    __shared__ __align__(16) _Float16 Pq[8][16][40]; // per-wave P^T staging [q][key]
    __shared__ float cbuf[8][16];
    __shared__ float lbuf[8][16];

    const int wave = threadIdx.x >> 5, lane = threadIdx.x & 31;
    const int lm = lane & 15, lh = lane >> 4;
    const int bh    = blockIdx.x >> 3;   // 0..127
    const int qblk  = blockIdx.x & 7;
    const int qbase = qblk * 128 + wave * 16;

    const _Float16* Qp = Q  + ((size_t)bh * SEQ + qbase) * HDIM;
    const _Float16* Kp = K  + (size_t)bh * SEQ * HDIM;
    const _Float16* Vp = Vt + (size_t)bh * HDIM * SEQ;

    v16h aq0 = ld_a16(&Qp[lm * HDIM +  0 + lh * 8]);
    v16h aq1 = ld_a16(&Qp[lm * HDIM + 32 + lh * 8]);

    v8f o0 = {}, o1 = {}, o2 = {}, o3 = {};
    float m8[8], l8[8];
    #pragma unroll
    for (int r = 0; r < 8; ++r) { m8[r] = -1e30f; l8[r] = 0.0f; }

    for (int kb = 0; kb < SEQ; kb += 32) {
        // scores: D[q,key] for two 16-key subtiles (lane = key column)
        v16h b0a = ld_b16(&Kp[(kb + lm) * HDIM +  0 + lh * 8]);
        v16h b0b = ld_b16(&Kp[(kb + lm) * HDIM + 32 + lh * 8]);
        v8f sA = {};
        sA = wmma_f16(aq0, b0a, sA);
        sA = wmma_f16(aq1, b0b, sA);
        v16h b1a = ld_b16(&Kp[(kb + 16 + lm) * HDIM +  0 + lh * 8]);
        v16h b1b = ld_b16(&Kp[(kb + 16 + lm) * HDIM + 32 + lh * 8]);
        v8f sB = {};
        sB = wmma_f16(aq0, b1a, sB);
        sB = wmma_f16(aq1, b1b, sB);

        // online softmax per query row r+8*lh; keys live across 16 lanes
        #pragma unroll
        for (int r = 0; r < 8; ++r) {
            float a = sA[r], bsc = sB[r];
            float tmax = fmaxf(a, bsc);
            tmax = fmaxf(tmax, __shfl_xor(tmax, 1, 32));
            tmax = fmaxf(tmax, __shfl_xor(tmax, 2, 32));
            tmax = fmaxf(tmax, __shfl_xor(tmax, 4, 32));
            tmax = fmaxf(tmax, __shfl_xor(tmax, 8, 32));
            float mnew = fmaxf(m8[r], tmax);
            float corr = __expf(m8[r] - mnew);
            float pA = __expf(a - mnew);
            float pB = __expf(bsc - mnew);
            float ps = pA + pB;
            ps += __shfl_xor(ps, 1, 32);
            ps += __shfl_xor(ps, 2, 32);
            ps += __shfl_xor(ps, 4, 32);
            ps += __shfl_xor(ps, 8, 32);
            l8[r] = l8[r] * corr + ps;
            m8[r] = mnew;
            const int qrow = r + 8 * lh;
            Pq[wave][qrow][lm]      = (_Float16)pA;
            Pq[wave][qrow][16 + lm] = (_Float16)pB;
            cbuf[wave][qrow] = corr;           // redundant identical writes
        }
        // rescale running output (lane column = query index lm)
        float cq = cbuf[wave][lm];
        #pragma unroll
        for (int r = 0; r < 8; ++r) { o0[r] *= cq; o1[r] *= cq; o2[r] *= cq; o3[r] *= cq; }

        // Ot[d,q] += V[d,k] @ P^T[k,q]  (K-dim = 32 keys)
        v16h bp  = ld_b16(&Pq[wave][lm][lh * 8]);
        v16h av0 = ld_a16(&Vp[( 0 + lm) * SEQ + kb + lh * 8]);
        v16h av1 = ld_a16(&Vp[(16 + lm) * SEQ + kb + lh * 8]);
        v16h av2 = ld_a16(&Vp[(32 + lm) * SEQ + kb + lh * 8]);
        v16h av3 = ld_a16(&Vp[(48 + lm) * SEQ + kb + lh * 8]);
        o0 = wmma_f16(av0, bp, o0);
        o1 = wmma_f16(av1, bp, o1);
        o2 = wmma_f16(av2, bp, o2);
        o3 = wmma_f16(av3, bp, o3);
    }

    #pragma unroll
    for (int r = 0; r < 8; ++r) lbuf[wave][r + 8 * lh] = l8[r];
    float linv = 1.0f / lbuf[wave][lm];

    const int b = bh >> 4, h = bh & (NHEADS - 1);
    const int srow = qbase + lm;
    const size_t outbase = ((size_t)(b * SEQ + srow)) * DIM + h * HDIM;
    #pragma unroll
    for (int r = 0; r < 8; ++r) {
        const int dr = r + 8 * lh;
        Out[outbase +  0 + dr] = (_Float16)(o0[r] * linv);
        Out[outbase + 16 + dr] = (_Float16)(o1[r] * linv);
        Out[outbase + 32 + dr] = (_Float16)(o2[r] * linv);
        Out[outbase + 48 + dr] = (_Float16)(o3[r] * linv);
    }
}

// ---------------- launcher ----------------
extern "C" void kernel_launch(void* const* d_in, const int* in_sizes, int n_in,
                              void* d_out, int out_size, void* d_ws, size_t ws_size,
                              hipStream_t stream) {
    const float* hidden = (const float*)d_in[0];
    const float* W_qkv  = (const float*)d_in[1];
    const float* b_qkv  = (const float*)d_in[2];
    const float* qn_w   = (const float*)d_in[3];
    const float* qn_b   = (const float*)d_in[4];
    const float* kn_w   = (const float*)d_in[5];
    const float* kn_b   = (const float*)d_in[6];
    const float* W_proj = (const float*)d_in[7];
    const float* b_proj = (const float*)d_in[8];
    const float* cosT   = (const float*)d_in[9];
    const float* sinT   = (const float*)d_in[10];
    float* out = (float*)d_out;

    char* ws = (char*)d_ws;
    size_t off = 0;
    auto take = [&](size_t bytes) -> char* {
        char* p = ws + off;
        off += (bytes + 255) & ~(size_t)255;
        return p;
    };
    _Float16* hidden_h = (_Float16*)take((size_t)ROWS * DIM * 2);       // 16 MB
    _Float16* wt_qkv   = (_Float16*)take((size_t)DIM * 3 * DIM * 2);    //  6 MB
    _Float16* wt_proj  = (_Float16*)take((size_t)DIM * DIM * 2);        //  2 MB
    _Float16* qkv_h    = (_Float16*)take((size_t)ROWS * 3 * DIM * 2);   // 48 MB
    _Float16* Qd       = (_Float16*)take((size_t)ROWS * DIM * 2);       // 16 MB
    _Float16* Kd       = (_Float16*)take((size_t)ROWS * DIM * 2);       // 16 MB
    _Float16* Vtd      = (_Float16*)take((size_t)ROWS * DIM * 2);       // 16 MB
    _Float16* attn_h   = (_Float16*)take((size_t)ROWS * DIM * 2);       // 16 MB

    na_cvt_f16<<<4096, 256, 0, stream>>>(hidden, hidden_h, ROWS * DIM);
    na_transpose_f16<<<(DIM * 3 * DIM) / 256, 256, 0, stream>>>(W_qkv, wt_qkv, DIM, 3 * DIM);
    na_transpose_f16<<<(DIM * DIM) / 256, 256, 0, stream>>>(W_proj, wt_proj, DIM, DIM);

    na_gemm_wmma<<<dim3(3 * DIM / 64, ROWS / 128), 256, 0, stream>>>(
        hidden_h, wt_qkv, b_qkv, qkv_h, nullptr, ROWS, 3 * DIM, DIM);

    na_norm_rope<<<(ROWS * NHEADS) / 8, 256, 0, stream>>>(
        qkv_h, qn_w, qn_b, kn_w, kn_b, cosT, sinT, Qd, Kd, Vtd);

    na_attention<<<BATCH * NHEADS * 8, 256, 0, stream>>>(Qd, Kd, Vtd, attn_h);

    na_gemm_wmma<<<dim3(DIM / 64, ROWS / 128), 256, 0, stream>>>(
        attn_h, wt_proj, b_proj, nullptr, out, ROWS, DIM, DIM);
}
